// MultiHeadedAttention_1297080123804
// MI455X (gfx1250) — compile-verified, matrix operand order checked by reading, not run on hardware
//
#include <hip/hip_runtime.h>
#include <hip/hip_bf16.h>

typedef __bf16 bf16;
typedef __attribute__((ext_vector_type(16))) __bf16 v16bf;
typedef __attribute__((ext_vector_type(8)))  __bf16 v8bf;
typedef __attribute__((ext_vector_type(8)))  float  v8f;
typedef __attribute__((ext_vector_type(4)))  float  v4f;
typedef __attribute__((ext_vector_type(4)))  int    v4i;

// Fixed problem dims (B,S,D,H) = (4,1024,1024,16)
constexpr int Bc  = 4;
constexpr int Sc  = 1024;
constexpr int Dc  = 1024;
constexpr int Hc  = 16;
constexpr int DKc = 64;
constexpr int Mc  = Bc * Sc;   // 4096

#define TM  128
#define TNN 128
#define TK  64
#define LDT 72      // 64 + 8 pad (bf16), 144B row stride: 16B-aligned, bank-conflict-dodging

// ---- optional gfx1250 async global->LDS copy path ---------------------------
#if defined(__has_builtin)
#if __has_builtin(__builtin_amdgcn_global_load_async_to_lds_b128) && \
    __has_builtin(__builtin_amdgcn_s_wait_asynccnt)
#define HAVE_ASYNC 1
#endif
#endif

#ifdef HAVE_ASYNC
typedef __attribute__((address_space(1))) char  gchar_t;
typedef __attribute__((address_space(3))) char  lchar_t;
typedef __attribute__((address_space(1))) v4i   gv4i_t;
typedef __attribute__((address_space(3))) v4i   lv4i_t;
#endif

__device__ __forceinline__ void cp_b128_to_lds(bf16* lds_dst, const bf16* gsrc) {
#ifdef HAVE_ASYNC
    gchar_t* g1 = (gchar_t*)(void*)gsrc;        // generic -> AS1 (addrspacecast)
    lchar_t* l1 = (lchar_t*)(void*)lds_dst;     // generic -> AS3
    __builtin_amdgcn_global_load_async_to_lds_b128((gv4i_t*)g1, (lv4i_t*)l1, 0, 0);
#else
    *(v8bf*)lds_dst = *(const v8bf*)gsrc;
#endif
}
__device__ __forceinline__ void cp_wait() {
#ifdef HAVE_ASYNC
    __builtin_amdgcn_s_wait_asynccnt(0);
#endif
}
// ----------------------------------------------------------------------------

__device__ __forceinline__ v8f wmma_bf16(v16bf a, v16bf b, v8f c) {
    return __builtin_amdgcn_wmma_f32_16x16x32_bf16(false, a, false, b, (short)0, c, false, false);
}

// 16x32 bf16 A-style fragment from LDS (also B = rows of an [N x K] matrix).
// lanes 0-15: row M=lane, v0..3 K={0..7}, v4..7 K={16..23}; lanes 16-31: +8.
__device__ __forceinline__ v16bf load_frag(const bf16* base, int row0, int k0, int lane) {
    int half = (lane >> 4) & 1;
    int r    = row0 + (lane & 15);
    const bf16* p = base + r * LDT + k0 + half * 8;
    union { v16bf v; v8bf h[2]; } u;
    u.h[0] = *(const v8bf*)(p);
    u.h[1] = *(const v8bf*)(p + 16);
    return u.v;
}

// ---------------------------------------------------------------------------
// C[M,N] = A[M,K] (f32) @ W[N,K]^T (f32) + bias, bf16 out permuted to (B,H,S,DK)
// Block tile 128x128, K-tile 64; 8 waves as 4(m) x 2(n), 32x64 per wave.
// ---------------------------------------------------------------------------
__global__ __launch_bounds__(256)
void gemm_qkv(const float* __restrict__ A, const float* __restrict__ W,
              const float* __restrict__ bias, bf16* __restrict__ out)
{
    __shared__ __align__(16) bf16 As[TM * LDT];
    __shared__ __align__(16) bf16 Ws[TNN * LDT];

    const int tid  = threadIdx.x;
    const int lane = tid & 31;
    const int wave = tid >> 5;
    const int wm   = wave >> 1;   // 0..3
    const int wn   = wave & 1;    // 0..1
    const int m0   = blockIdx.y * TM;
    const int n0   = blockIdx.x * TNN;

    // per-thread fixed staging coordinates: row r = (tid>>4)+16i, col = (tid&15)*4
    const int sr = tid >> 4, scc = (tid & 15) * 4;
    const float* Ap = A + (size_t)(m0 + sr) * Dc + scc;
    const float* Wp = W + (size_t)(n0 + sr) * Dc + scc;
    bf16* AsP = As + sr * LDT + scc;
    bf16* WsP = Ws + sr * LDT + scc;

    v8f acc[2][4];
    #pragma unroll
    for (int i = 0; i < 2; ++i)
        #pragma unroll
        for (int j = 0; j < 4; ++j) acc[i][j] = (v8f)0.0f;

    for (int k0 = 0; k0 < Dc; k0 += TK) {
        // phase 1: issue all 16 global b128 loads (batched, many in flight)
        v4f ta[8], tw[8];
        #pragma unroll
        for (int i = 0; i < 8; ++i) ta[i] = *(const v4f*)(Ap + k0 + (size_t)i * 16 * Dc);
        #pragma unroll
        for (int i = 0; i < 8; ++i) tw[i] = *(const v4f*)(Wp + k0 + (size_t)i * 16 * Dc);
        if (k0 + TK < Dc)
            __builtin_prefetch(Ap + k0 + TK, 0, 1);
        // phase 2: convert f32->bf16, store to LDS
        #pragma unroll
        for (int i = 0; i < 8; ++i) {
            bf16* d = AsP + i * 16 * LDT;
            d[0] = (bf16)ta[i].x; d[1] = (bf16)ta[i].y;
            d[2] = (bf16)ta[i].z; d[3] = (bf16)ta[i].w;
        }
        #pragma unroll
        for (int i = 0; i < 8; ++i) {
            bf16* d = WsP + i * 16 * LDT;
            d[0] = (bf16)tw[i].x; d[1] = (bf16)tw[i].y;
            d[2] = (bf16)tw[i].z; d[3] = (bf16)tw[i].w;
        }
        __syncthreads();

        #pragma unroll
        for (int kk = 0; kk < TK; kk += 32) {
            v16bf a0 = load_frag(As, wm * 32,      kk, lane);
            v16bf a1 = load_frag(As, wm * 32 + 16, kk, lane);
            #pragma unroll
            for (int jn = 0; jn < 4; ++jn) {
                v16bf bfr = load_frag(Ws, wn * 64 + jn * 16, kk, lane);
                acc[0][jn] = wmma_bf16(a0, bfr, acc[0][jn]);
                acc[1][jn] = wmma_bf16(a1, bfr, acc[1][jn]);
            }
        }
        __syncthreads();
    }

    // epilogue -> permuted bf16 (B,H,S,DK)
    const int halfl = lane >> 4, col = lane & 15;
    #pragma unroll
    for (int im = 0; im < 2; ++im)
        #pragma unroll
        for (int jn = 0; jn < 4; ++jn) {
            int mb = m0 + wm * 32 + im * 16 + halfl * 8;
            int n  = n0 + wn * 64 + jn * 16 + col;
            float bv = bias[n];
            int h = n >> 6, dk = n & 63;   // DK = 64
            #pragma unroll
            for (int i = 0; i < 8; ++i) {
                int m = mb + i;
                int b = m >> 10, s = m & 1023;   // S = 1024
                size_t idx = (((size_t)b * Hc + h) * Sc + s) * DKc + dk;
                out[idx] = (bf16)(acc[im][jn][i] + bv);
            }
        }
}

// ---------------------------------------------------------------------------
// out[M,N] (f32) = A[M,K] (bf16) @ W[N,K]^T (f32) + bias
// ---------------------------------------------------------------------------
__global__ __launch_bounds__(256)
void gemm_out(const bf16* __restrict__ A, const float* __restrict__ W,
              const float* __restrict__ bias, float* __restrict__ out)
{
    __shared__ __align__(16) bf16 As[TM * LDT];
    __shared__ __align__(16) bf16 Ws[TNN * LDT];

    const int tid  = threadIdx.x;
    const int lane = tid & 31;
    const int wave = tid >> 5;
    const int wm   = wave >> 1;
    const int wn   = wave & 1;
    const int m0   = blockIdx.y * TM;
    const int n0   = blockIdx.x * TNN;

    // A staging: bf16 8-elem chunks: row (tid>>3)+32i, col (tid&7)*8
    const int ar = tid >> 3, acc8 = (tid & 7) * 8;
    const bf16* Apb = A + (size_t)(m0 + ar) * Dc + acc8;
    bf16* AsP = As + ar * LDT + acc8;
    // W staging: f32 4-elem chunks
    const int sr = tid >> 4, scc = (tid & 15) * 4;
    const float* Wp = W + (size_t)(n0 + sr) * Dc + scc;
    bf16* WsP = Ws + sr * LDT + scc;

    v8f acc[2][4];
    #pragma unroll
    for (int i = 0; i < 2; ++i)
        #pragma unroll
        for (int j = 0; j < 4; ++j) acc[i][j] = (v8f)0.0f;

    for (int k0 = 0; k0 < Dc; k0 += TK) {
        #pragma unroll
        for (int i = 0; i < 4; ++i)
            cp_b128_to_lds(AsP + i * 32 * LDT, Apb + k0 + (size_t)i * 32 * Dc);
        v4f tw[8];
        #pragma unroll
        for (int i = 0; i < 8; ++i) tw[i] = *(const v4f*)(Wp + k0 + (size_t)i * 16 * Dc);
        #pragma unroll
        for (int i = 0; i < 8; ++i) {
            bf16* d = WsP + i * 16 * LDT;
            d[0] = (bf16)tw[i].x; d[1] = (bf16)tw[i].y;
            d[2] = (bf16)tw[i].z; d[3] = (bf16)tw[i].w;
        }
        cp_wait();
        __syncthreads();

        #pragma unroll
        for (int kk = 0; kk < TK; kk += 32) {
            v16bf a0 = load_frag(As, wm * 32,      kk, lane);
            v16bf a1 = load_frag(As, wm * 32 + 16, kk, lane);
            #pragma unroll
            for (int jn = 0; jn < 4; ++jn) {
                v16bf bfr = load_frag(Ws, wn * 64 + jn * 16, kk, lane);
                acc[0][jn] = wmma_bf16(a0, bfr, acc[0][jn]);
                acc[1][jn] = wmma_bf16(a1, bfr, acc[1][jn]);
            }
        }
        __syncthreads();
    }

    const int halfl = lane >> 4, col = lane & 15;
    #pragma unroll
    for (int im = 0; im < 2; ++im)
        #pragma unroll
        for (int jn = 0; jn < 4; ++jn) {
            int mb = m0 + wm * 32 + im * 16 + halfl * 8;
            int n  = n0 + wn * 64 + jn * 16 + col;
            float bv = bias[n];
            float* op = out + (size_t)mb * Dc + n;
            #pragma unroll
            for (int i = 0; i < 8; ++i)
                op[(size_t)i * Dc] = acc[im][jn][i] + bv;
        }
}

// ---------------------------------------------------------------------------
// Masked attention: x = (mask ? QK^T/8 : 0) @ V per (b,h); out -> (B,S,D) bf16
// ---------------------------------------------------------------------------
#define QT 128
#define KT 64

__global__ __launch_bounds__(256)
void attn(const bf16* __restrict__ Q, const bf16* __restrict__ Kb,
          const bf16* __restrict__ Vb, const int* __restrict__ mask,
          bf16* __restrict__ out)
{
    __shared__ __align__(16) bf16 Qs[QT * LDT];   // 128x64 q-tile
    __shared__ __align__(16) bf16 Ks[KT * LDT];   // 64x64 key-tile
    __shared__ __align__(16) bf16 Vt[DKc * LDT];  // transposed V [dk][s_k]
    __shared__ __align__(16) bf16 Ss[QT * LDT];   // masked bf16 scores [q][s_k]

    const int tid  = threadIdx.x;
    const int lane = tid & 31;
    const int wave = tid >> 5;
    const int wq   = wave >> 1;   // 0..3
    const int wn   = wave & 1;    // 0..1
    const int bh = blockIdx.y;
    const int b = bh >> 4, h = bh & 15;     // H = 16
    const int qt = blockIdx.x * QT;

    // staging coords: row (tid>>3)+32i, col (tid&7)*8
    const int sr = tid >> 3, scc = (tid & 7) * 8;
    const bf16* Qp = Q  + ((size_t)bh * Sc + qt + sr) * DKc + scc;
    const bf16* Kp = Kb + ((size_t)bh * Sc + sr) * DKc + scc;
    const bf16* Vp = Vb + ((size_t)bh * Sc + sr) * DKc + scc;

    // stage Q once (async if available)
    #pragma unroll
    for (int i = 0; i < 4; ++i)
        cp_b128_to_lds(Qs + (sr + 32 * i) * LDT + scc, Qp + (size_t)i * 32 * DKc);

    v8f xacc[2][2];
    #pragma unroll
    for (int i = 0; i < 2; ++i)
        #pragma unroll
        for (int j = 0; j < 2; ++j) xacc[i][j] = (v8f)0.0f;

    const int halfl = lane >> 4, col = lane & 15;

    for (int kt = 0; kt < Sc; kt += KT) {
        __syncthreads();   // prev pass-2 done reading Ks/Vt
        // stage K tile (async if available)
        #pragma unroll
        for (int i = 0; i < 2; ++i)
            cp_b128_to_lds(Ks + (sr + 32 * i) * LDT + scc,
                           Kp + (size_t)(kt + 32 * i) * DKc);
        // stage V tile transposed (register pass, batched loads then stores)
        v8bf tv[2];
        #pragma unroll
        for (int i = 0; i < 2; ++i)
            tv[i] = *(const v8bf*)(Vp + (size_t)(kt + 32 * i) * DKc);
        #pragma unroll
        for (int i = 0; i < 2; ++i)
            #pragma unroll
            for (int j = 0; j < 8; ++j)
                Vt[(scc + j) * LDT + sr + 32 * i] = tv[i][j];
        if (kt + KT < Sc)
            __builtin_prefetch(Kp + (size_t)(kt + KT) * DKc, 0, 1);
        cp_wait();
        __syncthreads();

        // ---- pass 1: S = Q @ K^T over dk (K-dim = 64)
        v8f sacc[2][2];
        #pragma unroll
        for (int i = 0; i < 2; ++i)
            #pragma unroll
            for (int j = 0; j < 2; ++j) sacc[i][j] = (v8f)0.0f;
        #pragma unroll
        for (int kk = 0; kk < DKc; kk += 32) {
            v16bf a0 = load_frag(Qs, wq * 32,      kk, lane);
            v16bf a1 = load_frag(Qs, wq * 32 + 16, kk, lane);
            v16bf b0 = load_frag(Ks, wn * 32,      kk, lane);
            v16bf b1 = load_frag(Ks, wn * 32 + 16, kk, lane);
            sacc[0][0] = wmma_bf16(a0, b0, sacc[0][0]);
            sacc[0][1] = wmma_bf16(a0, b1, sacc[0][1]);
            sacc[1][0] = wmma_bf16(a1, b0, sacc[1][0]);
            sacc[1][1] = wmma_bf16(a1, b1, sacc[1][1]);
        }
        // scale, mask (batched mask loads), store bf16 scores to LDS
        #pragma unroll
        for (int im = 0; im < 2; ++im)
            #pragma unroll
            for (int in = 0; in < 2; ++in) {
                int ql0 = wq * 32 + im * 16 + halfl * 8;
                int kl  = wn * 32 + in * 16 + col;
                const int* mp = mask + (size_t)(qt + ql0) * Sc + (kt + kl);
                int mv[8];
                #pragma unroll
                for (int i = 0; i < 8; ++i) mv[i] = mp[(size_t)i * Sc];
                #pragma unroll
                for (int i = 0; i < 8; ++i) {
                    float v = sacc[im][in][i] * 0.125f;   // 1/sqrt(64)
                    Ss[(ql0 + i) * LDT + kl] = (bf16)(mv[i] != 0 ? v : 0.0f);
                }
            }
        __syncthreads();

        // ---- pass 2: xacc += Ss[128x64] @ V  (B-fragments = rows of Vt)
        #pragma unroll
        for (int kk = 0; kk < KT; kk += 32) {
            v16bf a0 = load_frag(Ss, wq * 32,      kk, lane);
            v16bf a1 = load_frag(Ss, wq * 32 + 16, kk, lane);
            v16bf b0 = load_frag(Vt, wn * 32,      kk, lane);
            v16bf b1 = load_frag(Vt, wn * 32 + 16, kk, lane);
            xacc[0][0] = wmma_bf16(a0, b0, xacc[0][0]);
            xacc[0][1] = wmma_bf16(a0, b1, xacc[0][1]);
            xacc[1][0] = wmma_bf16(a1, b0, xacc[1][0]);
            xacc[1][1] = wmma_bf16(a1, b1, xacc[1][1]);
        }
    }

    // store x into (B,S,D) bf16 layout for the output projection
    #pragma unroll
    for (int im = 0; im < 2; ++im)
        #pragma unroll
        for (int in = 0; in < 2; ++in) {
            int q0 = qt + wq * 32 + im * 16 + halfl * 8;
            int dk = wn * 32 + in * 16 + col;
            bf16* op = out + ((size_t)b * Sc + q0) * Dc + h * DKc + dk;
            #pragma unroll
            for (int i = 0; i < 8; ++i)
                op[(size_t)i * Dc] = (bf16)xacc[im][in][i];
        }
}

// ---------------------------------------------------------------------------
extern "C" void kernel_launch(void* const* d_in, const int* in_sizes, int n_in,
                              void* d_out, int out_size, void* d_ws, size_t ws_size,
                              hipStream_t stream) {
    const float* query = (const float*)d_in[0];
    const float* key_  = (const float*)d_in[1];
    const float* value = (const float*)d_in[2];
    const int*   mask  = (const int*)d_in[3];
    const float* Wq = (const float*)d_in[4];
    const float* bq = (const float*)d_in[5];
    const float* Wk = (const float*)d_in[6];
    const float* bk = (const float*)d_in[7];
    const float* Wv = (const float*)d_in[8];
    const float* bv = (const float*)d_in[9];
    const float* Wo = (const float*)d_in[10];
    const float* bo = (const float*)d_in[11];
    float* out = (float*)d_out;

    bf16* qbuf = (bf16*)d_ws;              // 4 x (4096*1024) bf16 = 32 MB
    bf16* kbuf = qbuf + (size_t)Mc * Dc;
    bf16* vbuf = kbuf + (size_t)Mc * Dc;
    bf16* xbuf = vbuf + (size_t)Mc * Dc;

    dim3 blk(256);
    dim3 g1(Dc / TNN, Mc / TM);            // (8, 32)
    gemm_qkv<<<g1, blk, 0, stream>>>(query, Wq, bq, qbuf);
    gemm_qkv<<<g1, blk, 0, stream>>>(key_,  Wk, bk, kbuf);
    gemm_qkv<<<g1, blk, 0, stream>>>(value, Wv, bv, vbuf);

    dim3 g2(Sc / QT, Bc * Hc);             // (8, 64)
    attn<<<g2, blk, 0, stream>>>(qbuf, kbuf, vbuf, mask, xbuf);

    gemm_out<<<g1, blk, 0, stream>>>(xbuf, Wo, bo, out);
}
